// ConformerTransducer_25898652795600
// MI455X (gfx1250) — matrix operand order repaired and checked by reading, not compiled, not run on hardware
//
#include <hip/hip_runtime.h>
#include <hip/hip_bf16.h>

// ---------------------------------------------------------------------------
// Problem constants (from reference): B=4, T=200, U=100, V=1024,
// D_ENC=144, D_DEC=320, J=320.
// ---------------------------------------------------------------------------
#define RB 4
#define RT 200
#define RU 100        // U; U+1 = 101 rows in the u dimension
#define RV 1024
#define RDE 144
#define RDD 320
#define RJ 320

#define KTILES (RJ / 32)     // 10 K-steps of 32 (bf16 WMMA)
#define NTILES (RV / 16)     // 64 column tiles of 16
#define UTILES 7             // ceil(101/16)
#define MT 4                 // t-values (m-tiles) per workgroup: M = 64 rows
#define NT 4                 // n-tiles per wave (16 waves x 4 = 64 n-tiles)

typedef __attribute__((ext_vector_type(8)))  float          v8f;
typedef __attribute__((ext_vector_type(16))) __bf16         v16bf;
typedef __attribute__((ext_vector_type(8)))  unsigned short v8us;

union AccU  { v8f v; float f[8]; };
union AFrag { v16bf bf; v8us h[2]; };

__device__ __forceinline__ unsigned short f32_to_bf16_rne(float f) {
    unsigned int u = __float_as_uint(f);
    u += 0x7FFFu + ((u >> 16) & 1u);   // round to nearest even
    return (unsigned short)(u >> 16);
}

// ---------------------------------------------------------------------------
// Kernel 1/2: small fp32 projections  Y[row, j] = X[row,:] @ W[:, j] + bias[j]
// (0.16 GFLOP total — negligible next to the 53 GFLOP joint GEMM)
// ---------------------------------------------------------------------------
__global__ void rnnt_proj_kernel(const float* __restrict__ X,
                                 const float* __restrict__ W,
                                 const float* __restrict__ bias,
                                 float* __restrict__ Y, int rows, int K) {
    int id = blockIdx.x * blockDim.x + threadIdx.x;
    int total = rows * RJ;
    if (id >= total) return;
    int row = id / RJ;
    int j   = id - row * RJ;
    const float* x = X + row * K;
    float s = bias[j];
    for (int k = 0; k < K; ++k) s = fmaf(x[k], W[k * RJ + j], s);
    Y[id] = s;
}

// ---------------------------------------------------------------------------
// Kernel 3: stage W_out (f32 [320,1024]) into bf16 WMMA B-fragment order.
// Fragment layout (contiguous K order, same as the A reads): element e of lane L holds
// K = (L>>4)*16 + e, N = (L&15).  Flat index:
//   Bws[ ((kk*64 + nn)*32 + L)*16 + e ]
// ---------------------------------------------------------------------------
__global__ void rnnt_stageB_kernel(const float* __restrict__ W,
                                   unsigned short* __restrict__ Bws) {
    int id = blockIdx.x * blockDim.x + threadIdx.x;
    if (id >= KTILES * NTILES * 32 * 16) return;
    int e  = id & 15;
    int L  = (id >> 4) & 31;
    int nn = (id >> 9) & 63;
    int kk = id >> 15;
    int K = kk * 32 + (L >> 4) * 16 + e;
    int N = nn * 16 + (L & 15);
    Bws[id] = f32_to_bf16_rne(W[K * RV + N]);
}

// ---------------------------------------------------------------------------
// Kernel 4: fused joint + logits + log-softmax -> (blank_lp, emit_lp).
// One 512-thread WG per (b, 4 t-values, u-tile of 16 rows): M = 64 rows.
// 16 waves; wave w owns 4 n-tiles (cols [w*64, w*64+64)).
// Each B fragment is reused across 4 A fragments (4x less L2 traffic than an
// M=16 tile: ~0.9 GB total L2 reads of the 640 KB staged W).
// A (tanh tile, 64x320 bf16) lives in LDS; per wave: 160 WMMAs.
// ---------------------------------------------------------------------------
__global__ __launch_bounds__(512)
void rnnt_joint_kernel(const float* __restrict__ encp,
                       const float* __restrict__ decp,
                       const unsigned short* __restrict__ Bws,
                       const float* __restrict__ b_out,
                       const int* __restrict__ targets,
                       float* __restrict__ blankp,
                       float* __restrict__ emitp) {
    const int ut = blockIdx.x;           // 0..6
    const int t0 = blockIdx.y * MT;      // 0,4,...,196
    const int b  = blockIdx.z;           // 0..3
    const int tid  = threadIdx.x;
    const int lane = tid & 31;
    const int w    = tid >> 5;           // wave 0..15
    const int h    = lane >> 4;          // lane half
    const int c    = lane & 15;          // column within n-tile / A row id
    const int u0   = ut * 16;

    __shared__ unsigned short zlds[64 * RJ];   // 40 KB bf16 tanh tile
    __shared__ float wred[16][64];
    __shared__ float rowmax_s[64];
    __shared__ float rowlse_s[64];
    __shared__ int   tcol_s[16];

    // ---- build A tile: z[mt*16+m, j] = tanh(enc_p[b,t0+mt,j] + dec_p[b,u0+m,j])
    for (int i = tid; i < 64 * RJ; i += 512) {
        int row = i / RJ;
        int j   = i - row * RJ;
        int mt  = row >> 4;
        int m   = row & 15;
        int u = u0 + m; if (u > RU) u = RU;     // clamp for partial last tile
        float v = tanhf(encp[(b * RT + t0 + mt) * RJ + j] +
                        decp[(b * (RU + 1) + u) * RJ + j]);
        zlds[i] = f32_to_bf16_rne(v);
    }
    if (tid < 16) {
        int u = u0 + tid;
        tcol_s[tid] = (u < RU) ? targets[b * RU + u] : -1;
    }
    __syncthreads();

    // ---- WMMA main loop: 10 k-steps x (4 m-tiles x 4 n-tiles) per wave ----
    AccU acc[MT][NT];
    v8f zero = {0.f, 0.f, 0.f, 0.f, 0.f, 0.f, 0.f, 0.f};
#pragma unroll
    for (int mt = 0; mt < MT; ++mt)
#pragma unroll
        for (int j = 0; j < NT; ++j) acc[mt][j].v = zero;

    for (int kk = 0; kk < KTILES; ++kk) {
        // B fragments: 4 n-tiles, each reused across 4 A fragments
        v16bf bf[NT];
        const unsigned short* bp =
            Bws + (((kk * NTILES + w * NT) * 32) + lane) * 16;
#pragma unroll
        for (int j = 0; j < NT; ++j)
            bf[j] = *(const v16bf*)(bp + j * 32 * 16);
#pragma unroll
        for (int mt = 0; mt < MT; ++mt) {
            // A fragment (contiguous K order, matching staged B): lane half h
            // reads K [kk*32 + 16h, +16) of row m = c.
            const unsigned short* zr = zlds + (mt * 16 + c) * RJ;
            AFrag a;
            a.h[0] = *(const v8us*)(zr + kk * 32 + h * 16);
            a.h[1] = *(const v8us*)(zr + kk * 32 + h * 16 + 8);
#pragma unroll
            for (int j = 0; j < NT; ++j)
                acc[mt][j].v = __builtin_amdgcn_wmma_f32_16x16x32_bf16(
                    false, a.bf, false, bf[j], (short)0, acc[mt][j].v,
                    false, false);
        }
    }

    // ---- epilogue: per-row logsumexp over 1024 columns ----
    // acc[mt][j].f[r] = logits[t0+mt][row = r + 8h][col = w*64 + j*16 + c]
    float bias[NT];
#pragma unroll
    for (int j = 0; j < NT; ++j) bias[j] = b_out[w * 64 + j * 16 + c];

    // phase 1: row max (shfl within 16-lane halves, then cross-wave via LDS)
#pragma unroll
    for (int mt = 0; mt < MT; ++mt) {
#pragma unroll
        for (int r = 0; r < 8; ++r) {
            float m = -3.0e38f;
#pragma unroll
            for (int j = 0; j < NT; ++j)
                m = fmaxf(m, acc[mt][j].f[r] + bias[j]);
            for (int mask = 1; mask <= 8; mask <<= 1)
                m = fmaxf(m, __shfl_xor(m, mask, 32));
            if (c == 0) wred[w][mt * 16 + r + 8 * h] = m;
        }
    }
    __syncthreads();
    if (tid < 64) {
        float m = wred[0][tid];
        for (int ww = 1; ww < 16; ++ww) m = fmaxf(m, wred[ww][tid]);
        rowmax_s[tid] = m;
    }
    __syncthreads();

    // phase 2: row sum of exp(x - max)
#pragma unroll
    for (int mt = 0; mt < MT; ++mt) {
#pragma unroll
        for (int r = 0; r < 8; ++r) {
            int row = mt * 16 + r + 8 * h;
            float rm = rowmax_s[row];
            float s = 0.f;
#pragma unroll
            for (int j = 0; j < NT; ++j)
                s += __expf(acc[mt][j].f[r] + bias[j] - rm);
            for (int mask = 1; mask <= 8; mask <<= 1)
                s += __shfl_xor(s, mask, 32);
            if (c == 0) wred[w][row] = s;
        }
    }
    __syncthreads();
    if (tid < 64) {
        float s = 0.f;
        for (int ww = 0; ww < 16; ++ww) s += wred[ww][tid];
        rowlse_s[tid] = rowmax_s[tid] + __logf(s);
    }
    __syncthreads();

    // phase 3: scatter only the two columns the loss needs
#pragma unroll
    for (int mt = 0; mt < MT; ++mt) {
        int t = t0 + mt;
#pragma unroll
        for (int r = 0; r < 8; ++r) {
            int row = r + 8 * h;
            int u = u0 + row;
            if (u > RU) continue;
            float lse = rowlse_s[mt * 16 + row];
            int tc = tcol_s[row];
#pragma unroll
            for (int j = 0; j < NT; ++j) {
                int col = w * 64 + j * 16 + c;
                float lp = acc[mt][j].f[r] + bias[j] - lse;
                if (col == 0)
                    blankp[(b * RT + t) * (RU + 1) + u] = lp;
                if (col == tc)
                    emitp[(b * RT + t) * RU + u] = lp;
            }
        }
    }
}

// ---------------------------------------------------------------------------
// Kernel 5: RNN-T alpha recursion (inherently sequential scan), 4 batches
// in parallel; writes scalar loss = mean(-ll).
// ---------------------------------------------------------------------------
__global__ void rnnt_loss_kernel(const float* __restrict__ blankp,
                                 const float* __restrict__ emitp,
                                 const int* __restrict__ enc_len,
                                 const int* __restrict__ tgt_len,
                                 float* __restrict__ out) {
    __shared__ float alpha[RB][RU + 1];
    __shared__ float llsh[RB];
    int tid = threadIdx.x;
    if (tid < RB) {
        int b = tid;
        const float* bl = blankp + b * RT * (RU + 1);
        const float* em = emitp + b * RT * RU;
        int Tb = enc_len[b];
        int tl = tgt_len[b];
        float* a = alpha[b];
        a[0] = 0.f;
        for (int u = 1; u <= RU; ++u) a[u] = a[u - 1] + em[u - 1];
        float ll = 0.f;
        if (Tb == 1) ll = a[tl] + bl[tl];
        for (int t = 1; t < RT; ++t) {
            const float* blp = bl + (t - 1) * (RU + 1);
            const float* emt = em + t * RU;
            float prev = a[0] + blp[0];
            a[0] = prev;
            for (int u = 1; u <= RU; ++u) {
                float au = a[u] + blp[u];
                float x  = prev + emt[u - 1];
                float m  = fmaxf(au, x);
                float cur = m + log1pf(__expf(-fabsf(au - x)));
                a[u] = cur;
                prev = cur;
            }
            if (t == Tb - 1) ll = a[tl] + bl[t * (RU + 1) + tl];
        }
        llsh[b] = ll;
    }
    __syncthreads();
    if (tid == 0) {
        float s = 0.f;
        for (int b = 0; b < RB; ++b) s += llsh[b];
        out[0] = -s * (1.0f / RB);
    }
}

// ---------------------------------------------------------------------------
// Workspace layout (bytes, 128-aligned):
//   encp   @ 0        : 800*320 f32   = 1,024,000
//   decp   @ 1024000  : 404*320 f32   =   517,120
//   Bws    @ 1541120  : 327,680 u16   =   655,360
//   blankp @ 2196480  : 80,800 f32    =   323,200
//   emitp  @ 2519680  : 80,000 f32    =   320,000   (total ~2.84 MB)
// ---------------------------------------------------------------------------
extern "C" void kernel_launch(void* const* d_in, const int* in_sizes, int n_in,
                              void* d_out, int out_size, void* d_ws, size_t ws_size,
                              hipStream_t stream) {
    const float* enc_out   = (const float*)d_in[0];
    const float* dec_out   = (const float*)d_in[1];
    const float* W_enc     = (const float*)d_in[2];
    const float* b_enc     = (const float*)d_in[3];
    const float* W_dec     = (const float*)d_in[4];
    const float* b_dec     = (const float*)d_in[5];
    const float* W_out     = (const float*)d_in[6];
    const float* b_out     = (const float*)d_in[7];
    const int*   targets   = (const int*)d_in[8];
    const int*   enc_lens  = (const int*)d_in[9];
    const int*   tgt_lens  = (const int*)d_in[10];
    (void)in_sizes; (void)n_in; (void)out_size; (void)ws_size;

    char* ws = (char*)d_ws;
    float*          encp   = (float*)(ws + 0);
    float*          decp   = (float*)(ws + 1024000);
    unsigned short* Bws    = (unsigned short*)(ws + 1541120);
    float*          blankp = (float*)(ws + 2196480);
    float*          emitp  = (float*)(ws + 2519680);

    // enc projection: 800 rows x 320, K=144
    rnnt_proj_kernel<<<(RB * RT * RJ) / 256, 256, 0, stream>>>(
        enc_out, W_enc, b_enc, encp, RB * RT, RDE);
    // dec projection: 404 rows x 320, K=320
    rnnt_proj_kernel<<<(RB * (RU + 1) * RJ) / 256, 256, 0, stream>>>(
        dec_out, W_dec, b_dec, decp, RB * (RU + 1), RDD);
    // stage W_out into bf16 WMMA fragments
    rnnt_stageB_kernel<<<(KTILES * NTILES * 32 * 16) / 256, 256, 0, stream>>>(
        W_out, Bws);
    // fused joint + log-softmax (the 53 GFLOP WMMA GEMM), M=64 rows per WG
    dim3 grid(UTILES, RT / MT, RB);
    rnnt_joint_kernel<<<grid, 512, 0, stream>>>(
        encp, decp, Bws, b_out, targets, blankp, emitp);
    // alpha recursion -> scalar loss
    rnnt_loss_kernel<<<1, 64, 0, stream>>>(
        blankp, emitp, enc_lens, tgt_lens, (float*)d_out);
}
